// GAT_83167746719861
// MI455X (gfx1250) — compile-verified, matrix operand order checked by reading, not run on hardware
//
#include <hip/hip_runtime.h>
#include <hip/hip_bf16.h>

// Problem constants (match reference)
#define NN 100000
#define CC 128
#define HH 8
#define DD 16
#define EE 1600000
#define EPS_ 1e-5f
#define SLOPE 0.2f

typedef __attribute__((ext_vector_type(16))) _Float16 v16h;
typedef __attribute__((ext_vector_type(8)))  _Float16 v8h;
typedef __attribute__((ext_vector_type(8)))  float    v8f;

// ---------------------------------------------------------------------------
// WMMA helper (gfx1250 wave32, 16x16x32 f16 -> f32 accumulate)
// ---------------------------------------------------------------------------
__device__ __forceinline__ v8f wmma_f16(v16h a, v16h b, v8f c) {
  return __builtin_amdgcn_wmma_f32_16x16x32_f16(
      /*neg_a=*/false, a, /*neg_b=*/false, b,
      /*c_mod=*/(short)0, c, /*reuse_a=*/false, /*reuse_b=*/false);
}

// Fragment load from f32 source with inline f32->f16 convert (A operands that
// are read exactly once; conversion is overlapped with WMMA).
// ISA 16-bit 16x32 layout: lane<16 holds K=[k0..k0+7],[k0+16..k0+23];
// lane>=16 holds K=[k0+8..k0+15],[k0+24..k0+31].
__device__ __forceinline__ v16h load_frag_f32(const float* __restrict__ base,
                                              int row, int ld, int k0, int lane) {
  int half = (lane >> 4) & 1;
  const float* p = base + (size_t)row * ld + (k0 + half * 8);
  const float4* p4 = reinterpret_cast<const float4*>(p);
  float4 u0 = p4[0], u1 = p4[1];   // K = kb .. kb+7
  float4 u2 = p4[4], u3 = p4[5];   // K = kb+16 .. kb+23
  v16h a;
  a[0]  = (_Float16)u0.x; a[1]  = (_Float16)u0.y; a[2]  = (_Float16)u0.z; a[3]  = (_Float16)u0.w;
  a[4]  = (_Float16)u1.x; a[5]  = (_Float16)u1.y; a[6]  = (_Float16)u1.z; a[7]  = (_Float16)u1.w;
  a[8]  = (_Float16)u2.x; a[9]  = (_Float16)u2.y; a[10] = (_Float16)u2.z; a[11] = (_Float16)u2.w;
  a[12] = (_Float16)u3.x; a[13] = (_Float16)u3.y; a[14] = (_Float16)u3.z; a[15] = (_Float16)u3.w;
  return a;
}

// Fragment load from pre-converted f16 source: two 16B vector loads, no VALU.
__device__ __forceinline__ v16h load_frag_f16(const _Float16* __restrict__ base,
                                              int row, int ld, int k0, int lane) {
  int half = (lane >> 4) & 1;
  const _Float16* p = base + (size_t)row * ld + (k0 + half * 8);
  v8h lo = *(const v8h*)p;
  v8h hi = *(const v8h*)(p + 16);
  return __builtin_shufflevector(lo, hi, 0, 1, 2, 3, 4, 5, 6, 7,
                                 8, 9, 10, 11, 12, 13, 14, 15);
}

// ---------------------------------------------------------------------------
// Monotonic float<->uint mapping for atomicMax on signed floats
// ---------------------------------------------------------------------------
__device__ __forceinline__ unsigned f2mono(float f) {
  unsigned u = __float_as_uint(f);
  return (u & 0x80000000u) ? ~u : (u | 0x80000000u);
}
__device__ __forceinline__ float mono2f(unsigned u) {
  return (u & 0x80000000u) ? __uint_as_float(u & 0x7fffffffu)
                           : __uint_as_float(~u);
}

// ---------------------------------------------------------------------------
// f32 -> f16 convert (weights, one-time)
// ---------------------------------------------------------------------------
__global__ void k_cvt(const float* __restrict__ s, _Float16* __restrict__ d, int n) {
  int i = blockIdx.x * blockDim.x + threadIdx.x;
  if (i < n) d[i] = (_Float16)s[i];
}

// Init: gat accumulator = gat_bias (folds bias add), softmax state
__global__ void k_init(float* __restrict__ gat_out, const float* __restrict__ gat_bias,
                       unsigned* __restrict__ mmono, float* __restrict__ denom) {
  int i = blockIdx.x * blockDim.x + threadIdx.x;
  if (i < NN * CC) gat_out[i] = gat_bias[i & (CC - 1)];
  if (i < NN * HH) { mmono[i] = 0u; denom[i] = 0.f; }
}

// ---------------------------------------------------------------------------
// x_l = x0 @ lin_l_w^T ; x_r = x0 @ lin_r_w^T  (blockIdx.y selects which)
// Outputs stored as f16 -> halves the edge-phase gather traffic.
// ---------------------------------------------------------------------------
__global__ __launch_bounds__(128) void k_gemm_lr(const float* __restrict__ x0,
                                                 const _Float16* __restrict__ wlh,
                                                 const _Float16* __restrict__ wrh,
                                                 _Float16* __restrict__ xlh,
                                                 _Float16* __restrict__ xrh) {
  int lane = threadIdx.x & 31;
  int wave = blockIdx.x * 4 + (threadIdx.x >> 5);
  const _Float16* W = blockIdx.y ? wrh : wlh;
  _Float16* Y = blockIdx.y ? xrh : xlh;
  int row0 = wave * 16;
  int m = lane & 15, half = lane >> 4;
  int rA = min(row0 + m, NN - 1);

  v8f acc[8];
#pragma unroll
  for (int t = 0; t < 8; ++t)
#pragma unroll
    for (int v = 0; v < 8; ++v) acc[t][v] = 0.f;

#pragma unroll
  for (int kt = 0; kt < 4; ++kt) {
    v16h a = load_frag_f32(x0, rA, CC, kt * 32, lane);
#pragma unroll
    for (int t = 0; t < 8; ++t) {
      v16h b = load_frag_f16(W, t * 16 + m, CC, kt * 32, lane);
      acc[t] = wmma_f16(a, b, acc[t]);
    }
  }
#pragma unroll
  for (int t = 0; t < 8; ++t) {
    int col = t * 16 + m;
#pragma unroll
    for (int v = 0; v < 8; ++v) {
      int row = row0 + half * 8 + v;
      if (row < NN) Y[(size_t)row * CC + col] = (_Float16)acc[t][v];
    }
  }
}

// ---------------------------------------------------------------------------
// Edge pass 1: logits[e,h] = att[h] . leaky_relu(x_l[src,h,:] + x_r[dst,h,:])
// ---------------------------------------------------------------------------
__global__ void k_edge_logits(const int* __restrict__ edges,
                              const _Float16* __restrict__ xlh,
                              const _Float16* __restrict__ xrh,
                              const float* __restrict__ att,
                              float* __restrict__ logits,
                              unsigned* __restrict__ mmono) {
  int i = blockIdx.x * blockDim.x + threadIdx.x;
  if (i >= EE * HH) return;
  int e = i >> 3, h = i & 7;
  int src = edges[e], dst = edges[EE + e];
  const _Float16* pl = xlh + (size_t)src * CC + h * DD;
  const _Float16* pr = xrh + (size_t)dst * CC + h * DD;
  v8h l0 = *(const v8h*)pl, l1 = *(const v8h*)(pl + 8);
  v8h r0 = *(const v8h*)pr, r1 = *(const v8h*)(pr + 8);
  float acc = 0.f;
#pragma unroll
  for (int d = 0; d < 8; ++d) {
    float v0 = (float)l0[d] + (float)r0[d];
    v0 = v0 > 0.f ? v0 : v0 * SLOPE;
    float v1 = (float)l1[d] + (float)r1[d];
    v1 = v1 > 0.f ? v1 : v1 * SLOPE;
    acc += att[h * DD + d] * v0 + att[h * DD + 8 + d] * v1;
  }
  logits[i] = acc;
  atomicMax(mmono + (size_t)dst * HH + h, f2mono(acc));
}

// Edge pass 2: ex = exp(logit - max); denom += ex (store ex over logits)
__global__ void k_edge_exp(const int* __restrict__ edges,
                           float* __restrict__ logits,
                           const unsigned* __restrict__ mmono,
                           float* __restrict__ denom) {
  int i = blockIdx.x * blockDim.x + threadIdx.x;
  if (i >= EE * HH) return;
  int e = i >> 3, h = i & 7;
  int dst = edges[EE + e];
  float mx = mono2f(mmono[(size_t)dst * HH + h]);
  float ex = __expf(logits[i] - mx);
  logits[i] = ex;
  atomicAdd(denom + (size_t)dst * HH + h, ex);
}

// Edge pass 3: out[dst,h,:] += (ex/denom) * x_l[src,h,:]
__global__ void k_edge_scatter(const int* __restrict__ edges,
                               const _Float16* __restrict__ xlh,
                               const float* __restrict__ logits,
                               const float* __restrict__ denom,
                               float* __restrict__ gat_out) {
  int i = blockIdx.x * blockDim.x + threadIdx.x;
  if (i >= EE * HH) return;
  int e = i >> 3, h = i & 7;
  int src = edges[e], dst = edges[EE + e];
  float al = logits[i] / (denom[(size_t)dst * HH + h] + 1e-16f);
  const _Float16* pl = xlh + (size_t)src * CC + h * DD;
  v8h l0 = *(const v8h*)pl, l1 = *(const v8h*)(pl + 8);
  float* po = gat_out + (size_t)dst * CC + h * DD;
#pragma unroll
  for (int d = 0; d < 8; ++d) {
    atomicAdd(po + d,     al * (float)l0[d]);
    atomicAdd(po + 8 + d, al * (float)l1[d]);
  }
}

// ---------------------------------------------------------------------------
// x3 = LayerNorm(x1 @ W_w^T + x0) * g1 + b1   (x1 = gat_out incl. bias)
// Emits f32 x3 (residual) and f16 x3h (MLP A operand).
// ---------------------------------------------------------------------------
__global__ __launch_bounds__(128) void k_gemm_res_ln(const float* __restrict__ x1,
                                                     const _Float16* __restrict__ wwh,
                                                     const float* __restrict__ x0,
                                                     const float* __restrict__ g1,
                                                     const float* __restrict__ b1,
                                                     float* __restrict__ x3,
                                                     _Float16* __restrict__ x3h) {
  int lane = threadIdx.x & 31;
  int wave = blockIdx.x * 4 + (threadIdx.x >> 5);
  int row0 = wave * 16;
  int m = lane & 15, half = lane >> 4;
  int rA = min(row0 + m, NN - 1);

  v8f acc[8];
#pragma unroll
  for (int t = 0; t < 8; ++t)
#pragma unroll
    for (int v = 0; v < 8; ++v) acc[t][v] = 0.f;

#pragma unroll
  for (int kt = 0; kt < 4; ++kt) {
    v16h a = load_frag_f32(x1, rA, CC, kt * 32, lane);
#pragma unroll
    for (int t = 0; t < 8; ++t) {
      v16h b = load_frag_f16(wwh, t * 16 + m, CC, kt * 32, lane);
      acc[t] = wmma_f16(a, b, acc[t]);
    }
  }

  float s[8], sq[8];
#pragma unroll
  for (int v = 0; v < 8; ++v) { s[v] = 0.f; sq[v] = 0.f; }
#pragma unroll
  for (int t = 0; t < 8; ++t) {
    int col = t * 16 + m;
#pragma unroll
    for (int v = 0; v < 8; ++v) {
      int rr = min(row0 + half * 8 + v, NN - 1);
      float x = acc[t][v] + x0[(size_t)rr * CC + col];
      acc[t][v] = x;
      s[v] += x; sq[v] += x * x;
    }
  }
#pragma unroll
  for (int mask = 1; mask < 16; mask <<= 1)
#pragma unroll
    for (int v = 0; v < 8; ++v) {
      s[v]  += __shfl_xor(s[v],  mask, 32);
      sq[v] += __shfl_xor(sq[v], mask, 32);
    }
  float mu[8], rs[8];
#pragma unroll
  for (int v = 0; v < 8; ++v) {
    mu[v] = s[v] * (1.f / CC);
    float var = sq[v] * (1.f / CC) - mu[v] * mu[v];
    rs[v] = rsqrtf(var + EPS_);
  }
#pragma unroll
  for (int t = 0; t < 8; ++t) {
    int col = t * 16 + m;
    float g = g1[col], bb = b1[col];
#pragma unroll
    for (int v = 0; v < 8; ++v) {
      int row = row0 + half * 8 + v;
      if (row < NN) {
        float y = (acc[t][v] - mu[v]) * rs[v] * g + bb;
        x3[(size_t)row * CC + col] = y;
        x3h[(size_t)row * CC + col] = (_Float16)y;
      }
    }
  }
}

// ---------------------------------------------------------------------------
// Fused MLP: h = relu(x3@w1^T + b1m) (16x256 f16 tile staged in LDS),
//            x4 = h@w2^T + b2m + x3, out = LN(x4)*g2 + b2
// ---------------------------------------------------------------------------
__global__ __launch_bounds__(128) void k_mlp(const _Float16* __restrict__ x3h,
                                             const float* __restrict__ x3,
                                             const _Float16* __restrict__ w1h,
                                             const float* __restrict__ b1m,
                                             const _Float16* __restrict__ w2h,
                                             const float* __restrict__ b2m,
                                             const float* __restrict__ g2,
                                             const float* __restrict__ b2v,
                                             float* __restrict__ out) {
  __shared__ _Float16 hbuf[4 * 16 * 256];   // 32 KB: per-wave 16x256 f16 tile
  int lane = threadIdx.x & 31, wid = threadIdx.x >> 5;
  int wave = blockIdx.x * 4 + wid;
  int row0 = wave * 16;
  int m = lane & 15, half = lane >> 4;
  int rA = min(row0 + m, NN - 1);
  _Float16* hb = hbuf + wid * 16 * 256;

  // GEMM1 in two 128-wide column halves to cap register pressure
#pragma unroll
  for (int ht = 0; ht < 2; ++ht) {
    v8f acc[8];
#pragma unroll
    for (int t = 0; t < 8; ++t)
#pragma unroll
      for (int v = 0; v < 8; ++v) acc[t][v] = 0.f;
#pragma unroll
    for (int kt = 0; kt < 4; ++kt) {
      v16h a = load_frag_f16(x3h, rA, CC, kt * 32, lane);
#pragma unroll
      for (int t = 0; t < 8; ++t) {
        v16h b = load_frag_f16(w1h, (ht * 8 + t) * 16 + m, CC, kt * 32, lane);
        acc[t] = wmma_f16(a, b, acc[t]);
      }
    }
    // bias + relu -> LDS as f16 (A operand of GEMM2)
#pragma unroll
    for (int t = 0; t < 8; ++t) {
      int col = (ht * 8 + t) * 16 + m;
      float bi = b1m[col];
#pragma unroll
      for (int v = 0; v < 8; ++v) {
        float hv = acc[t][v] + bi;
        hv = hv > 0.f ? hv : 0.f;
        hb[(half * 8 + v) * 256 + col] = (_Float16)hv;
      }
    }
  }
  __syncthreads();

  // GEMM2: K=256 from LDS (vector ds_load_b128 pairs), 8 output tiles
  v8f acc2[8];
#pragma unroll
  for (int t = 0; t < 8; ++t)
#pragma unroll
    for (int v = 0; v < 8; ++v) acc2[t][v] = 0.f;
#pragma unroll
  for (int kt = 0; kt < 8; ++kt) {
    const _Float16* hp = hb + m * 256 + kt * 32 + half * 8;
    v8h lo = *(const v8h*)hp;
    v8h hi = *(const v8h*)(hp + 16);
    v16h a = __builtin_shufflevector(lo, hi, 0, 1, 2, 3, 4, 5, 6, 7,
                                     8, 9, 10, 11, 12, 13, 14, 15);
#pragma unroll
    for (int t = 0; t < 8; ++t) {
      v16h b = load_frag_f16(w2h, t * 16 + m, 2 * CC, kt * 32, lane);
      acc2[t] = wmma_f16(a, b, acc2[t]);
    }
  }

  // + b2m + x3 residual, then LN(g2, b2v)
  float s[8], sq[8];
#pragma unroll
  for (int v = 0; v < 8; ++v) { s[v] = 0.f; sq[v] = 0.f; }
#pragma unroll
  for (int t = 0; t < 8; ++t) {
    int col = t * 16 + m;
    float bi = b2m[col];
#pragma unroll
    for (int v = 0; v < 8; ++v) {
      int rr = min(row0 + half * 8 + v, NN - 1);
      float x = acc2[t][v] + bi + x3[(size_t)rr * CC + col];
      acc2[t][v] = x;
      s[v] += x; sq[v] += x * x;
    }
  }
#pragma unroll
  for (int mask = 1; mask < 16; mask <<= 1)
#pragma unroll
    for (int v = 0; v < 8; ++v) {
      s[v]  += __shfl_xor(s[v],  mask, 32);
      sq[v] += __shfl_xor(sq[v], mask, 32);
    }
  float mu[8], rs[8];
#pragma unroll
  for (int v = 0; v < 8; ++v) {
    mu[v] = s[v] * (1.f / CC);
    float var = sq[v] * (1.f / CC) - mu[v] * mu[v];
    rs[v] = rsqrtf(var + EPS_);
  }
#pragma unroll
  for (int t = 0; t < 8; ++t) {
    int col = t * 16 + m;
    float g = g2[col], bb = b2v[col];
#pragma unroll
    for (int v = 0; v < 8; ++v) {
      int row = row0 + half * 8 + v;
      if (row < NN)
        out[(size_t)row * CC + col] = (acc2[t][v] - mu[v]) * rs[v] * g + bb;
    }
  }
}

// ---------------------------------------------------------------------------
extern "C" void kernel_launch(void* const* d_in, const int* in_sizes, int n_in,
                              void* d_out, int out_size, void* d_ws, size_t ws_size,
                              hipStream_t stream) {
  (void)in_sizes; (void)n_in; (void)out_size; (void)ws_size;
  const float* x0       = (const float*)d_in[0];
  const int*   edges    = (const int*)d_in[1];
  const float* lin_l_w  = (const float*)d_in[2];
  const float* lin_r_w  = (const float*)d_in[3];
  const float* att      = (const float*)d_in[4];
  const float* gat_bias = (const float*)d_in[5];
  const float* W_w      = (const float*)d_in[6];
  const float* g1       = (const float*)d_in[7];
  const float* b1       = (const float*)d_in[8];
  const float* mlp_w1   = (const float*)d_in[9];
  const float* mlp_b1   = (const float*)d_in[10];
  const float* mlp_w2   = (const float*)d_in[11];
  const float* mlp_b2   = (const float*)d_in[12];
  const float* g2       = (const float*)d_in[13];
  const float* b2       = (const float*)d_in[14];
  float* out = (float*)d_out;

  // ---- workspace layout: f32/u32 regions first, then f16 regions ----
  char* wsb = (char*)d_ws;
  size_t off = 0;
  float* gat_out = (float*)(wsb + off); off += (size_t)NN * CC * 4;
  float* x3      = (float*)(wsb + off); off += (size_t)NN * CC * 4;
  float* logits  = (float*)(wsb + off); off += (size_t)EE * HH * 4;
  unsigned* mmono= (unsigned*)(wsb + off); off += (size_t)NN * HH * 4;
  float* denom   = (float*)(wsb + off); off += (size_t)NN * HH * 4;
  _Float16* xlh  = (_Float16*)(wsb + off); off += (size_t)NN * CC * 2;
  _Float16* xrh  = (_Float16*)(wsb + off); off += (size_t)NN * CC * 2;
  _Float16* x3h  = (_Float16*)(wsb + off); off += (size_t)NN * CC * 2;
  _Float16* wlh  = (_Float16*)(wsb + off); off += (size_t)CC * CC * 2;
  _Float16* wrh  = (_Float16*)(wsb + off); off += (size_t)CC * CC * 2;
  _Float16* wwh  = (_Float16*)(wsb + off); off += (size_t)CC * CC * 2;
  _Float16* w1h  = (_Float16*)(wsb + off); off += (size_t)2 * CC * CC * 2;
  _Float16* w2h  = (_Float16*)(wsb + off); off += (size_t)2 * CC * CC * 2;

  const int gemm_blocks = (((NN + 15) / 16) + 3) / 4;      // 4 waves/block
  const int edge_blocks = (EE * HH + 255) / 256;
  const int init_blocks = (NN * CC + 255) / 256;
  const int wcvt_blocks = (CC * CC + 255) / 256;

  // one-time weight conversion to f16 (removes per-wave cvt VALU)
  k_cvt<<<wcvt_blocks, 256, 0, stream>>>(lin_l_w, wlh, CC * CC);
  k_cvt<<<wcvt_blocks, 256, 0, stream>>>(lin_r_w, wrh, CC * CC);
  k_cvt<<<wcvt_blocks, 256, 0, stream>>>(W_w,     wwh, CC * CC);
  k_cvt<<<2 * wcvt_blocks, 256, 0, stream>>>(mlp_w1, w1h, 2 * CC * CC);
  k_cvt<<<2 * wcvt_blocks, 256, 0, stream>>>(mlp_w2, w2h, 2 * CC * CC);

  k_init<<<init_blocks, 256, 0, stream>>>(gat_out, gat_bias, mmono, denom);
  dim3 glr(gemm_blocks, 2);
  k_gemm_lr<<<glr, 128, 0, stream>>>(x0, wlh, wrh, xlh, xrh);
  k_edge_logits<<<edge_blocks, 256, 0, stream>>>(edges, xlh, xrh, att, logits, mmono);
  k_edge_exp<<<edge_blocks, 256, 0, stream>>>(edges, logits, mmono, denom);
  k_edge_scatter<<<edge_blocks, 256, 0, stream>>>(edges, xlh, logits, denom, gat_out);
  k_gemm_res_ln<<<gemm_blocks, 128, 0, stream>>>(gat_out, wwh, x0, g1, b1, x3, x3h);
  k_mlp<<<gemm_blocks, 128, 0, stream>>>(x3h, x3, w1h, mlp_b1, w2h, mlp_b2, g2, b2, out);
}
